// WindowAttention_12369505812842
// MI455X (gfx1250) — compile-verified, matrix operand order checked by reading, not run on hardware
//
#include <hip/hip_runtime.h>

typedef __attribute__((ext_vector_type(4)))  __bf16 v4bf;
typedef __attribute__((ext_vector_type(8)))  __bf16 v8bf;
typedef __attribute__((ext_vector_type(16))) __bf16 v16bf;
typedef __attribute__((ext_vector_type(8)))  float  v8f;

#define DIM     192
#define HEADS   6
#define HD      32
#define NTOK    49
#define NPAD    64
#define NWIN    64
#define NBW     4096
#define QKV_OUT 576
#define SCALE   0.17677669529663687f   // 1/sqrt(32)

// ---- workspace layout (bytes) ----
#define WS_QKVW   0                    // 576*192 bf16  = 221184 B
#define WS_PROJW  221184               // 192*192 bf16  =  73728 B
#define WS_BMD    294912               // 64*6*4*4*32*8 f32 = 6291456 B (C/D-layout bias+mask)

// ---------------------------------------------------------------------------
// A/B operand loader for v_wmma_*_16x16x32_bf16.
// Lane L: row/col index r = L&15, half g = L>>4.
// elements 0..7  -> k = 8g .. 8g+7          (one 16B load)
// elements 8..15 -> k = 16+8g .. 16+8g+7    (one 16B load)
// ---------------------------------------------------------------------------
__device__ __forceinline__ v16bf load_ab(const __bf16* base, int stride, int r, int g) {
  const __bf16* p = base + r * stride + 8 * g;
  v8bf lo = *(const v8bf*)(p);
  v8bf hi = *(const v8bf*)(p + 16);
  return __builtin_shufflevector(lo, hi, 0,1,2,3,4,5,6,7,8,9,10,11,12,13,14,15);
}

__device__ __forceinline__ v8f wmma_bf(v16bf a, v16bf b, v8f c) {
  return __builtin_amdgcn_wmma_f32_16x16x32_bf16(false, a, false, b, (short)0, c, false, false);
}

// ---------------------------------------------------------------------------
// Prep 1: fp32 weights -> bf16
// ---------------------------------------------------------------------------
__global__ void prep_weights(const float* __restrict__ qkv_w,
                             const float* __restrict__ proj_w,
                             __bf16* __restrict__ qkvw_bf,
                             __bf16* __restrict__ projw_bf) {
  int i = blockIdx.x * 256 + threadIdx.x;
  if (i < QKV_OUT * DIM) qkvw_bf[i] = (__bf16)qkv_w[i];
  if (i < DIM * DIM)     projw_bf[i] = (__bf16)proj_w[i];
}

// ---------------------------------------------------------------------------
// Prep 2: bias+mask table pre-swizzled into the WMMA C/D register layout:
//   bmd[w][h][mt][nt][lane][v] ,  lane in [0,32), v in [0,8)
//   row  m = mt*16 + v + 8*(lane>>4)   (clamped to 48 for pad rows)
//   col  j = nt*16 + (lane&15)         (-1e9 for j >= 49 -> exp underflows to 0,
//                                       which also neutralizes K/V padding rows)
// ---------------------------------------------------------------------------
__global__ void prep_bmd(const int* __restrict__ rpi,
                         const float* __restrict__ mask,
                         const float* __restrict__ rpb,
                         float* __restrict__ bmd) {
  int idx = blockIdx.x * 256 + threadIdx.x;
  if (idx >= NWIN * HEADS * 4 * 4 * 32 * 8) return;
  int v    = idx & 7;
  int lane = (idx >> 3) & 31;
  int nt   = (idx >> 8) & 3;
  int mt   = (idx >> 10) & 3;
  int h    = (idx >> 12) % HEADS;
  int w    = idx / (4096 * HEADS);
  int m = mt * 16 + v + 8 * (lane >> 4);
  int i = m < NTOK ? m : NTOK - 1;
  int j = nt * 16 + (lane & 15);
  float val = -1e9f;
  if (j < NTOK)
    val = mask[(w * NTOK + i) * NTOK + j] + rpb[rpi[i * NTOK + j] * HEADS + h];
  bmd[idx] = val;
}

// ---------------------------------------------------------------------------
// Fused window attention: 1 block = 1 window (8 wave32s)
// ---------------------------------------------------------------------------
__global__ void __launch_bounds__(256)
win_attn(const float* __restrict__ x,
         const float* __restrict__ qkv_b,
         const float* __restrict__ proj_b,
         const __bf16* __restrict__ qkvw,
         const __bf16* __restrict__ projw,
         const float* __restrict__ bmd,
         float* __restrict__ out) {
  extern __shared__ char smem[];
  __bf16* xw  = (__bf16*)(smem);            // [64][192]  (phase3: reused as attn-out)
  __bf16* qs  = (__bf16*)(smem + 24576);    // [6][64][32]  Q * scale
  __bf16* ksh = (__bf16*)(smem + 49152);    // [6][64][32]  K
  __bf16* vt  = (__bf16*)(smem + 73728);    // [6][32][64]  V transposed
  __bf16* ps  = (__bf16*)(smem + 98304);    // [8 waves][16][64]  P strips

  // wave id, forced provably-uniform so tile loops & branches go scalar
  const int wvu  = __builtin_amdgcn_readfirstlane(threadIdx.x >> 5);
  const int lane = threadIdx.x & 31;
  const int g    = lane >> 4;
  const int r    = lane & 15;
  const int bw   = blockIdx.x;

  // warm caches for the (L2-resident) bf16 weights while x streams in
  __builtin_prefetch(qkvw + (size_t)threadIdx.x * 432, 0, 3);
  __builtin_prefetch(projw + (size_t)threadIdx.x * 144, 0, 3);

  // ---- phase 0: stream x window -> bf16 LDS (pad rows 49..63 with zero) ----
  {
    const float4* xsrc = (const float4*)(x + (size_t)bw * (NTOK * DIM));
    for (int i4 = threadIdx.x; i4 < (NPAD * DIM) / 4; i4 += 256) {
      v4bf hv;
      if (i4 < (NTOK * DIM) / 4) {
        float4 f = xsrc[i4];
        hv[0] = (__bf16)f.x; hv[1] = (__bf16)f.y;
        hv[2] = (__bf16)f.z; hv[3] = (__bf16)f.w;
      } else {
        hv[0] = hv[1] = hv[2] = hv[3] = (__bf16)0.0f;
      }
      *(v4bf*)(xw + i4 * 4) = hv;
    }
  }
  __syncthreads();

  // ---- phase 1: QKV GEMM [64x192]@[192x576] -> Q/K/V in LDS ----
  // Padding rows (m >= 49) need only be FINITE, not zero: padded S columns get
  // -1e9 from bmd (=> P exactly 0), padded O/S rows are never stored.
  for (int t = wvu; t < 4 * 36; t += 8) {
    const int mt = t / 36, nt = t % 36;       // scalar (t is uniform)
    v8f acc = {};
#pragma unroll
    for (int kk = 0; kk < 6; ++kk) {
      v16bf a = load_ab(xw   + (mt * 16) * DIM + kk * 32, DIM, r, g);
      v16bf b = load_ab(qkvw + (nt * 16) * DIM + kk * 32, DIM, r, g);
      acc = wmma_bf(a, b, acc);
    }
    const int idx3 = nt / 12;                 // scalar: which of q/k/v
    const int c0   = (nt % 12) * 16;          // scalar column base within 192
    const int c = c0 + r, h = c >> 5, d = c & 31;
    float bo = qkv_b[idx3 * DIM + c];
    if (idx3 == 0) {
#pragma unroll
      for (int v = 0; v < 8; ++v) {
        int m = mt * 16 + v + 8 * g;
        qs[(h * NPAD + m) * HD + d] = (__bf16)((acc[v] + bo) * SCALE);
      }
    } else if (idx3 == 1) {
#pragma unroll
      for (int v = 0; v < 8; ++v) {
        int m = mt * 16 + v + 8 * g;
        ksh[(h * NPAD + m) * HD + d] = (__bf16)(acc[v] + bo);
      }
    } else {
#pragma unroll
      for (int v = 0; v < 8; ++v) {
        int m = mt * 16 + v + 8 * g;
        vt[(h * HD + d) * NPAD + m] = (__bf16)(acc[v] + bo);
      }
    }
  }
  __syncthreads();

  // ---- phase 2: attention, 24 strips = (head, 16-row m-tile) ----
  const int w_in_img = bw & (NWIN - 1);
  for (int strip = wvu; strip < HEADS * 4; strip += 8) {
    const int h = strip >> 2, mt = strip & 3;   // scalar

    // S strip [16 x 64] = Q_strip @ K^T  (4 WMMAs, K-dim = 32)
    v16bf aq = load_ab(qs + (h * NPAD + mt * 16) * HD, HD, r, g);
    v8f s[4];
#pragma unroll
    for (int nt = 0; nt < 4; ++nt) {
      v8f z = {};
      v16bf bk = load_ab(ksh + (h * NPAD + nt * 16) * HD, HD, r, g);
      s[nt] = wmma_bf(aq, bk, z);
    }

    // + bias + mask, pre-swizzled in C/D layout -> 8 coalesced b128 loads
    const float* bstrip =
        bmd + ((size_t)(((w_in_img * HEADS + h) * 4 + mt) * 4) * 32 + lane) * 8;
#pragma unroll
    for (int nt = 0; nt < 4; ++nt) {
      const float4* bl = (const float4*)(bstrip + nt * 32 * 8);
      float4 b0 = bl[0], b1 = bl[1];
      s[nt][0] += b0.x; s[nt][1] += b0.y; s[nt][2] += b0.z; s[nt][3] += b0.w;
      s[nt][4] += b1.x; s[nt][5] += b1.y; s[nt][6] += b1.z; s[nt][7] += b1.w;
    }

    // softmax along the 64 columns (each row lives in one 16-lane half)
#pragma unroll
    for (int v = 0; v < 8; ++v) {
      float mx = fmaxf(fmaxf(s[0][v], s[1][v]), fmaxf(s[2][v], s[3][v]));
      for (int msk = 1; msk < 16; msk <<= 1) mx = fmaxf(mx, __shfl_xor(mx, msk, 32));
      float e0 = __expf(s[0][v] - mx), e1 = __expf(s[1][v] - mx);
      float e2 = __expf(s[2][v] - mx), e3 = __expf(s[3][v] - mx);
      float sum = e0 + e1 + e2 + e3;
      for (int msk = 1; msk < 16; msk <<= 1) sum += __shfl_xor(sum, msk, 32);
      float inv = 1.0f / sum;
      s[0][v] = e0 * inv; s[1][v] = e1 * inv; s[2][v] = e2 * inv; s[3][v] = e3 * inv;
    }

    // P strip -> LDS (C/D layout -> row-major bf16)
    __bf16* pw = ps + wvu * 16 * NPAD;
#pragma unroll
    for (int v = 0; v < 8; ++v) {
      int rr = v + 8 * g;
      pw[rr * NPAD +      r] = (__bf16)s[0][v];
      pw[rr * NPAD + 16 + r] = (__bf16)s[1][v];
      pw[rr * NPAD + 32 + r] = (__bf16)s[2][v];
      pw[rr * NPAD + 48 + r] = (__bf16)s[3][v];
    }
    asm volatile("s_wait_dscnt 0" ::: "memory");   // own-wave LDS RAW

    // O strip [16 x 32] = P @ V   (2 d-tiles x 2 k-steps)
#pragma unroll
    for (int dt = 0; dt < 2; ++dt) {
      v8f o = {};
#pragma unroll
      for (int k2 = 0; k2 < 2; ++k2) {
        v16bf ap = load_ab(pw + k2 * 32, NPAD, r, g);
        v16bf bv = load_ab(vt + (h * HD + dt * 16) * NPAD + k2 * 32, NPAD, r, g);
        o = wmma_bf(ap, bv, o);
      }
#pragma unroll
      for (int v = 0; v < 8; ++v) {
        int m = mt * 16 + v + 8 * g;
        xw[m * DIM + h * HD + dt * 16 + r] = (__bf16)o[v];  // reuse xw as attn-out
      }
    }
  }
  __syncthreads();

  // ---- phase 3: proj GEMM [49x192]@[192x192] + bias -> out ----
  float* outw = out + (size_t)bw * (NTOK * DIM);
  for (int t = wvu; t < 4 * 12; t += 8) {
    const int mt = t / 12, nt = t % 12;        // scalar
    v8f acc = {};
#pragma unroll
    for (int kk = 0; kk < 6; ++kk) {
      v16bf a = load_ab(xw    + (mt * 16) * DIM + kk * 32, DIM, r, g);
      v16bf b = load_ab(projw + (nt * 16) * DIM + kk * 32, DIM, r, g);
      acc = wmma_bf(a, b, acc);
    }
    int o = nt * 16 + r;
    float bo = proj_b[o];
    if (mt < 3) {
      // rows m = mt*16 + v + 8g <= 47 < 49: no guard needed (scalar branch)
#pragma unroll
      for (int v = 0; v < 8; ++v) {
        int m = mt * 16 + v + 8 * g;
        outw[m * DIM + o] = acc[v] + bo;
      }
    } else {
#pragma unroll
      for (int v = 0; v < 8; ++v) {
        int m = 48 + v + 8 * g;
        if (m < NTOK) outw[m * DIM + o] = acc[v] + bo;   // only row 48 survives
      }
    }
  }
}

// ---------------------------------------------------------------------------
extern "C" void kernel_launch(void* const* d_in, const int* in_sizes, int n_in,
                              void* d_out, int out_size, void* d_ws, size_t ws_size,
                              hipStream_t stream) {
  const float* x      = (const float*)d_in[0];
  const int*   rpi    = (const int*)  d_in[1];
  const float* mask   = (const float*)d_in[2];
  const float* qkv_w  = (const float*)d_in[3];
  const float* qkv_b  = (const float*)d_in[4];
  const float* proj_w = (const float*)d_in[5];
  const float* proj_b = (const float*)d_in[6];
  const float* rpb    = (const float*)d_in[7];
  float* out = (float*)d_out;

  char* ws = (char*)d_ws;
  __bf16* qkvw_bf  = (__bf16*)(ws + WS_QKVW);
  __bf16* projw_bf = (__bf16*)(ws + WS_PROJW);
  float*  bmd      = (float*) (ws + WS_BMD);

  prep_weights<<<(QKV_OUT * DIM + 255) / 256, 256, 0, stream>>>(qkv_w, proj_w, qkvw_bf, projw_bf);
  prep_bmd<<<(NWIN * HEADS * 4 * 4 * 32 * 8 + 255) / 256, 256, 0, stream>>>(rpi, mask, rpb, bmd);

  // 112 KB dynamic LDS per block (320 KB per WGP on CDNA5)
  win_attn<<<NBW, 256, 114688, stream>>>(x, qkv_b, proj_b, qkvw_bf, projw_bf, bmd, out);
}
